// myYOLO_7808250544643
// MI455X (gfx1250) — matrix operand order repaired and checked by reading, not compiled, not run on hardware
//
#include <hip/hip_runtime.h>
#include <hip/hip_bf16.h>
#include <math.h>

typedef __attribute__((ext_vector_type(16))) _Float16 v16h;
typedef __attribute__((ext_vector_type(8)))  float    v8f;
typedef __attribute__((ext_vector_type(4)))  float    v4f;

#define NUM_CLASSES 20
#define CONF_THRESH 0.01f
#define NMS_THRESH  0.5f
#define IOU_EPS     1e-28f

// ---------------------------------------------------------------------------
// Kernel 1: pack w_pred (25x512 f32, row-major) into the WMMA A-operand
// (16-bit A, 16x32) VGPR layout, O padded to 32 with zeros.
// wpack layout: [o_tile(2)][k_step(16)][lane(32)][p(16)] halfs.
// Per ISA 7.12.2 (16-bit A 16x32): lane L -> row M=L&15, h=L>>4;
// half p (v=p>>1, odd=p&1): K = (v<4) ? 8h+2v+odd : 16+8h+2(v-4)+odd.
// ---------------------------------------------------------------------------
__global__ void __launch_bounds__(256)
pack_weights(const float* __restrict__ w, _Float16* __restrict__ wpack) {
  int tid = blockIdx.x * blockDim.x + threadIdx.x;   // 0..1023 = (t*16+s)*32+L
  int t = tid >> 9;          // o-tile 0..1
  int s = (tid >> 5) & 15;   // k-step 0..15
  int L = tid & 31;
  int m = L & 15, h = L >> 4;
  int o = t * 16 + m;
  _Float16* dst = wpack + (size_t)tid * 16;
#pragma unroll
  for (int p = 0; p < 16; ++p) {
    int v = p >> 1, odd = p & 1;
    int k = (v < 4) ? (8 * h + 2 * v + odd) : (16 + 8 * h + 2 * (v - 4) + odd);
    int c = s * 32 + k;
    float val = (o < 25) ? w[o * 512 + c] : 0.0f;
    dst[p] = (_Float16)val;
  }
}

// ---------------------------------------------------------------------------
// Kernel 2: projection GEMM via WMMA.
// pred[b][n][o] = sum_c feat[b][c][n] * w[o][c]   (bias added in postprocess)
// One wave per 16-wide spatial tile; 2 o-tiles (o=0..15, 16..31).
// K=512 -> 16 steps of K=32, processed 2 per loop trip into 4 independent
// WMMA accumulator chains (avoids back-to-back same-C WMMA stalls).
// feat is streamed once -> non-temporal loads (keep L2 for pred).
// ---------------------------------------------------------------------------
__global__ void __launch_bounds__(256)
yolo_head_gemm(const float* __restrict__ feat, const _Float16* __restrict__ wpack,
               float* __restrict__ pred) {
  const int lane = threadIdx.x & 31;
  const int wave = threadIdx.x >> 5;
  const int tile = blockIdx.x * 8 + wave;        // 0..4095
  const int b    = tile >> 6;                    // batch
  const int n0   = (tile & 63) * 16;             // spatial tile base
  const int h    = lane >> 4, l = lane & 15;
  const float* fbase = feat + (size_t)b * 512 * 1024 + n0 + l;

  v8f acc0a = {}, acc1a = {};    // even K-steps
  v8f acc0b = {}, acc1b = {};    // odd  K-steps
#pragma unroll 1
  for (int s = 0; s < 16; s += 2) {
    // A operands for both steps first (critical path into first WMMA)
    const v16h a00 = *(const v16h*)(wpack + ((size_t)(0 * 16 + s    ) * 32 + lane) * 16);
    const v16h a01 = *(const v16h*)(wpack + ((size_t)(1 * 16 + s    ) * 32 + lane) * 16);
    const v16h a10 = *(const v16h*)(wpack + ((size_t)(0 * 16 + s + 1) * 32 + lane) * 16);
    const v16h a11 = *(const v16h*)(wpack + ((size_t)(1 * 16 + s + 1) * 32 + lane) * 16);

    // B operands (32x16 f16, KxN): lane (h,l) holds col N=l, K = 8h+2v(+1) (+8 for v>=4)
    const int c0 = s * 32 + 8 * h;
    v16h bmat0, bmat1;
#pragma unroll
    for (int v = 0; v < 8; ++v) {
      float f0 = __builtin_nontemporal_load(fbase + (size_t)(c0 + (v >= 4 ? 8 : 0) + 2 * v    ) * 1024);
      float f1 = __builtin_nontemporal_load(fbase + (size_t)(c0 + (v >= 4 ? 8 : 0) + 2 * v + 1) * 1024);
      bmat0[2 * v]     = (_Float16)f0;
      bmat0[2 * v + 1] = (_Float16)f1;
    }
#pragma unroll
    for (int v = 0; v < 8; ++v) {
      float f0 = __builtin_nontemporal_load(fbase + (size_t)(c0 + 32 + (v >= 4 ? 8 : 0) + 2 * v    ) * 1024);
      float f1 = __builtin_nontemporal_load(fbase + (size_t)(c0 + 32 + (v >= 4 ? 8 : 0) + 2 * v + 1) * 1024);
      bmat1[2 * v]     = (_Float16)f0;
      bmat1[2 * v + 1] = (_Float16)f1;
    }
    if (s < 14)  // global_prefetch_b8 for next loop trip
      __builtin_prefetch(fbase + (size_t)(c0 + 64) * 1024, 0, 1);

    acc0a = __builtin_amdgcn_wmma_f32_16x16x32_f16(false, a00, false, bmat0,
                                                   (short)0, acc0a, false, false);
    acc1a = __builtin_amdgcn_wmma_f32_16x16x32_f16(false, a01, false, bmat0,
                                                   (short)0, acc1a, false, false);
    acc0b = __builtin_amdgcn_wmma_f32_16x16x32_f16(false, a10, false, bmat1,
                                                   (short)0, acc0b, false, false);
    acc1b = __builtin_amdgcn_wmma_f32_16x16x32_f16(false, a11, false, bmat1,
                                                   (short)0, acc1b, false, false);
  }
  v8f acc0 = acc0a + acc0b;
  v8f acc1 = acc1a + acc1b;

  // D layout: lane (h,l) -> n = n0+l, o = r + 8h (acc0) / 16 + r + 8h (acc1)
  float* prow = pred + ((size_t)b * 1024 + n0 + l) * 32 + 8 * h;
  v4f lo0 = {acc0[0], acc0[1], acc0[2], acc0[3]};
  v4f hi0 = {acc0[4], acc0[5], acc0[6], acc0[7]};
  v4f lo1 = {acc1[0], acc1[1], acc1[2], acc1[3]};
  v4f hi1 = {acc1[4], acc1[5], acc1[6], acc1[7]};
  *(v4f*)(prow)      = lo0;
  *(v4f*)(prow + 4)  = hi0;
  *(v4f*)(prow + 16) = lo1;
  *(v4f*)(prow + 20) = hi1;
}

// ---------------------------------------------------------------------------
// Kernel 3: per-cell postprocess (bias, sigmoid, softmax, decode, argmax).
// One thread per (b, n); 65536 threads total.
// ---------------------------------------------------------------------------
__global__ void __launch_bounds__(256)
yolo_postprocess(const float* __restrict__ pred, const float* __restrict__ bias,
                 float* __restrict__ boxes, float* __restrict__ scores,
                 float* __restrict__ cls_out, float* __restrict__ offbox) {
  int gid = blockIdx.x * blockDim.x + threadIdx.x;   // 0..65535
  const float* p = pred + (size_t)gid * 32;
  float v[25];
#pragma unroll
  for (int o = 0; o < 25; ++o) v[o] = p[o] + bias[o];

  float conf = 1.0f / (1.0f + __expf(-v[0]));

  float m = v[1];
#pragma unroll
  for (int c = 1; c < NUM_CLASSES; ++c) m = fmaxf(m, v[1 + c]);
  float e[NUM_CLASSES];
  float sum = 0.0f;
#pragma unroll
  for (int c = 0; c < NUM_CLASSES; ++c) { e[c] = __expf(v[1 + c] - m); sum += e[c]; }
  int ci = 0; float best = e[0];
#pragma unroll
  for (int c = 1; c < NUM_CLASSES; ++c) if (e[c] > best) { best = e[c]; ci = c; }
  float score = conf * best / sum;

  int n = gid & 1023;
  float gx = (float)(n & 31), gy = (float)(n >> 5);
  float cx = (1.0f / (1.0f + __expf(-v[21])) + gx) * 32.0f;
  float cy = (1.0f / (1.0f + __expf(-v[22])) + gy) * 32.0f;
  float bw = __expf(v[23]), bh = __expf(v[24]);
  const float inv = 1.0f / 1024.0f;
  float x1 = fminf(fmaxf((cx - 0.5f * bw) * inv, 0.0f), 1.0f);
  float y1 = fminf(fmaxf((cy - 0.5f * bh) * inv, 0.0f), 1.0f);
  float x2 = fminf(fmaxf((cx + 0.5f * bw) * inv, 0.0f), 1.0f);
  float y2 = fminf(fmaxf((cy + 0.5f * bh) * inv, 0.0f), 1.0f);

  boxes[gid * 4 + 0] = x1;
  boxes[gid * 4 + 1] = y1;
  boxes[gid * 4 + 2] = x2;
  boxes[gid * 4 + 3] = y2;
  scores[gid]  = score;
  cls_out[gid] = (float)ci;

  float off = (float)ci * 2.0f;   // class-offset boxes for class-aware NMS
  offbox[gid * 4 + 0] = x1 + off;
  offbox[gid * 4 + 1] = y1 + off;
  offbox[gid * 4 + 2] = x2 + off;
  offbox[gid * 4 + 3] = y2 + off;
}

// ---------------------------------------------------------------------------
// Kernel 4: per-batch NMS. 1 block (1024 threads = 32 waves) per batch.
// Bitonic sort (score desc, index payload) in LDS, then a 1024-step
// suppression sweep over the sorted list. 28 KB LDS << 320 KB/WGP.
// ---------------------------------------------------------------------------
__global__ void __launch_bounds__(1024)
yolo_nms(const float* __restrict__ scores, const float* __restrict__ offbox,
         float* __restrict__ keep_out) {
  __shared__ float ss[1024];
  __shared__ int   si[1024];
  __shared__ float sb[1024 * 4];
  __shared__ int   sk[1024];
  const int tid = threadIdx.x;
  const int b   = blockIdx.x;

  ss[tid] = scores[b * 1024 + tid];
  si[tid] = tid;
  __syncthreads();

  // bitonic sort, descending by score
  for (int k = 2; k <= 1024; k <<= 1) {
    for (int j = k >> 1; j > 0; j >>= 1) {
      int ixj = tid ^ j;
      if (ixj > tid) {
        float a = ss[tid], c = ss[ixj];
        bool descSeg = ((tid & k) == 0);
        bool doswap  = descSeg ? (a < c) : (a > c);
        if (doswap) {
          ss[tid] = c; ss[ixj] = a;
          int t = si[tid]; si[tid] = si[ixj]; si[ixj] = t;
        }
      }
      __syncthreads();
    }
  }

  const int src = si[tid];
  float bx1 = offbox[(b * 1024 + src) * 4 + 0];
  float by1 = offbox[(b * 1024 + src) * 4 + 1];
  float bx2 = offbox[(b * 1024 + src) * 4 + 2];
  float by2 = offbox[(b * 1024 + src) * 4 + 3];
  sb[tid * 4 + 0] = bx1;
  sb[tid * 4 + 1] = by1;
  sb[tid * 4 + 2] = bx2;
  sb[tid * 4 + 3] = by2;
  float area = (bx2 - bx1) * (by2 - by1);
  sk[tid] = (ss[tid] > CONF_THRESH) ? 1 : 0;
  __syncthreads();

  for (int i = 0; i < 1024; ++i) {
    if (sk[i]) {
      if (tid > i && sk[tid]) {
        float px1 = sb[i * 4 + 0], py1 = sb[i * 4 + 1];
        float px2 = sb[i * 4 + 2], py2 = sb[i * 4 + 3];
        float xx1 = fmaxf(px1, bx1), yy1 = fmaxf(py1, by1);
        float xx2 = fminf(px2, bx2), yy2 = fminf(py2, by2);
        float w  = fmaxf(IOU_EPS, xx2 - xx1);
        float hh = fmaxf(IOU_EPS, yy2 - yy1);
        float inter = w * hh;
        float parea = (px2 - px1) * (py2 - py1);
        float iou = inter / (parea + area - inter);
        if (iou > NMS_THRESH) sk[tid] = 0;
      }
    }
    __syncthreads();
  }

  keep_out[b * 1024 + src] = sk[tid] ? 1.0f : 0.0f;
}

// ---------------------------------------------------------------------------
extern "C" void kernel_launch(void* const* d_in, const int* in_sizes, int n_in,
                              void* d_out, int out_size, void* d_ws, size_t ws_size,
                              hipStream_t stream) {
  const float* feat = (const float*)d_in[0];   // (64, 512, 32, 32)
  const float* w    = (const float*)d_in[1];   // (25, 512)
  const float* bias = (const float*)d_in[2];   // (25,)

  char* ws = (char*)d_ws;
  _Float16* wpack = (_Float16*)ws;                                   // 32 KB
  float*    pred  = (float*)(ws + 32 * 1024);                        // 8 MB
  float*    offb  = (float*)(ws + 32 * 1024 + (size_t)64 * 1024 * 32 * 4); // 1 MB

  float* out    = (float*)d_out;
  float* boxes  = out;                         // 64*1024*4
  float* scores = out + 262144;                // 64*1024
  float* cls    = out + 262144 + 65536;        // 64*1024
  float* keep   = out + 262144 + 2 * 65536;    // 64*1024

  pack_weights    <<<4,   256, 0, stream>>>(w, wpack);
  yolo_head_gemm  <<<512, 256, 0, stream>>>(feat, wpack, pred);
  yolo_postprocess<<<256, 256, 0, stream>>>(pred, bias, boxes, scores, cls, offb);
  yolo_nms        <<<64, 1024, 0, stream>>>(scores, offb, keep);
}
